// ConvFlow_51273319579937
// MI455X (gfx1250) — matrix operand order repaired and checked by reading, not compiled, not run on hardware
//
#include <hip/hip_runtime.h>
#include <hip/hip_bf16.h>
#include <math.h>

// ---------------------------------------------------------------------------
// Problem constants (from reference)
// ---------------------------------------------------------------------------
#define BATCH   16
#define C_IN    4
#define HALF    2
#define TLEN    4096
#define FILTER  512
#define KS      5
#define PAD     2
#define NLAYERS 4
#define NBINS   10
#define PROJ_CH 58            // HALF * (3*NBINS - 1)

// Conv-as-GEMM tiling
#define M_BLK   128           // channels per workgroup
#define N_BLK   64            // time positions per workgroup
#define KC_STEP 32            // input channels per K chunk
#define NKC     (FILTER / KC_STEP)  // 16 K chunks
#define LDS_ROWS (N_BLK + KS - 1)   // 68 (halo for 5-tap conv)
#define LDS_RS   40                 // row stride in halves; 80B rows -> conflict-free b128 gathers
#define STAGE_ELEMS (KC_STEP * LDS_ROWS)           // 2176
#define NSTAGE ((STAGE_ELEMS + 255) / 256)         // 9 per-thread staging slots

typedef __attribute__((ext_vector_type(16))) __bf16 v16bf;
typedef __attribute__((ext_vector_type(8)))  float  v8f;

union Frag {
    v16bf v;
    uint4 q[2];
    unsigned short h[16];
};

__device__ __forceinline__ float bf2f(unsigned short h) {
    unsigned int u = ((unsigned int)h) << 16;
    float f;
    __builtin_memcpy(&f, &u, 4);
    return f;
}
__device__ __forceinline__ unsigned short f2bf(float f) {
    unsigned int u;
    __builtin_memcpy(&u, &f, 4);
    unsigned int lsb = (u >> 16) & 1u;
    u += 0x7fffu + lsb;            // round-to-nearest-even
    return (unsigned short)(u >> 16);
}

// ---------------------------------------------------------------------------
// 1) Repack conv weights fp32 -> bf16 in WMMA A-fragment order.
// Layout: ((((layer*32 + mtile)*16 + kc)*5 + tap)*32 + lane)*16 + j
// This code's A fragment (contiguous k, same as B): lane<16 holds row M=lane, K=0..15;
// lane>=16 holds same row, K=16..31.
// ---------------------------------------------------------------------------
__global__ void cf_pack_w(const float* __restrict__ conv_w,
                          unsigned short* __restrict__ pw) {
    long long idx = (long long)blockIdx.x * blockDim.x + threadIdx.x;
    if (idx >= (long long)NLAYERS * 32 * 16 * KS * 32 * 16) return;
    int j    = (int)(idx & 15);
    int lane = (int)((idx >> 4) & 31);
    long long r = idx >> 9;
    int tap = (int)(r % 5); r /= 5;
    int kc  = (int)(r & 15); r >>= 4;
    int mt  = (int)(r & 31); r >>= 5;
    int l   = (int)r;
    int m = mt * 16 + (lane & 15);
    int c = kc * 32 + j + ((lane >> 4) << 4);
    float v = conv_w[(((long long)(l * FILTER + m)) * FILTER + c) * KS + tap];
    pw[idx] = f2bf(v);
}

// ---------------------------------------------------------------------------
// 2) Pre conv (pointwise 2 -> 512), no mask on its input per reference.
// ---------------------------------------------------------------------------
__global__ void cf_pre(const float* __restrict__ x,
                       const float* __restrict__ pre_w,
                       const float* __restrict__ pre_b,
                       unsigned short* __restrict__ act) {
    long long idx = (long long)blockIdx.x * blockDim.x + threadIdx.x; // (b*512+f)*T + t
    int t = (int)(idx & (TLEN - 1));
    int f = (int)((idx >> 12) & (FILTER - 1));
    int b = (int)(idx >> 21);
    float a0 = x[((long long)b * C_IN + 0) * TLEN + t];
    float a1 = x[((long long)b * C_IN + 1) * TLEN + t];
    float v = pre_w[f * 2 + 0] * a0 + pre_w[f * 2 + 1] * a1 + pre_b[f];
    act[idx] = f2bf(v);
}

// ---------------------------------------------------------------------------
// 3) 5-tap conv as WMMA GEMM with double-buffered LDS staging.
// One workgroup: 128 ch x 64 t; 8 waves, each a 32x32 output
// (2x2 tiles of v_wmma_f32_16x16x32_bf16). Next K chunk's global loads are
// issued before the current chunk's 20 WMMAs; LDS commit happens after, into
// the opposite buffer, so a single barrier per iteration both publishes
// buffer kc+1 and protects buffer kc. Mask folded into staging (reference
// masks each conv layer's input). kc+2 chunk is prefetched toward L2.
// ---------------------------------------------------------------------------
__global__ __launch_bounds__(256) void cf_conv_wmma(
        const unsigned short* __restrict__ act,
        const unsigned short* __restrict__ pw,
        const float* __restrict__ bias,
        const float* __restrict__ mask,
        float* __restrict__ out,
        int layer) {
    __shared__ unsigned short lds[2][LDS_ROWS * LDS_RS];

    const int tid   = threadIdx.x;
    const int b     = blockIdx.z;
    const int t0    = blockIdx.x * N_BLK;
    const int mBlk  = blockIdx.y * M_BLK;
    const int lane  = tid & 31;
    const int wv    = tid >> 5;
    const int waveM = wv >> 1;        // 0..3
    const int waveN = wv & 1;         // 0..1
    const int lhi   = lane >> 4;      // half-wave select
    const int llo   = lane & 15;

    v8f acc00 = {}, acc01 = {}, acc10 = {}, acc11 = {};

    const unsigned short* actB  = act  + (long long)b * FILTER * TLEN;
    const float*          maskB = mask + (long long)b * TLEN;

    // --- prologue: stage kc = 0 into buffer 0 ---
    #pragma unroll
    for (int s = 0; s < NSTAGE; ++s) {
        int i = tid + s * 256;
        if (i < STAGE_ELEMS) {
            int k = i / LDS_ROWS;
            int n = i - k * LDS_ROWS;
            int t = t0 + n - PAD;
            float v = 0.0f;
            if (t >= 0 && t < TLEN)
                v = bf2f(actB[(long long)k * TLEN + t]) * maskB[t];
            lds[0][n * LDS_RS + k] = f2bf(v);
        }
    }
    __syncthreads();

    for (int kc = 0; kc < NKC; ++kc) {
        const unsigned short* ldsCur = lds[kc & 1];
        const int kcn = kc + 1;

        // --- issue next chunk's global loads into registers (overlap WMMA) ---
        unsigned short reg[NSTAGE];
        if (kcn < NKC) {
            #pragma unroll
            for (int s = 0; s < NSTAGE; ++s) {
                int i = tid + s * 256;
                unsigned short rv = 0;
                if (i < STAGE_ELEMS) {
                    int k = i / LDS_ROWS;
                    int n = i - k * LDS_ROWS;
                    int t = t0 + n - PAD;
                    if (t >= 0 && t < TLEN)
                        rv = f2bf(bf2f(actB[(long long)(kcn * KC_STEP + k) * TLEN + t])
                                  * maskB[t]);
                }
                reg[s] = rv;
            }
            // pull kc+2 toward L2 (global_prefetch_b8)
            if (kc + 2 < NKC) {
                int k = tid / LDS_ROWS;
                int n = tid - k * LDS_ROWS;
                int t = min(max(t0 + n - PAD, 0), TLEN - 1);
                __builtin_prefetch(actB + (long long)((kc + 2) * KC_STEP + k) * TLEN + t,
                                   0, 1);
            }
        }

        // --- compute current chunk: 5 taps x 2x2 WMMA tiles ---
        #pragma unroll
        for (int tap = 0; tap < KS; ++tap) {
            Frag a0, a1, b0, b1;
            // A fragments: pre-packed, coalesced 32B per lane, L2-resident.
            const int mt0 = blockIdx.y * (M_BLK / 16) + waveM * 2;
            const uint4* p0 = (const uint4*)(pw +
                (((((long long)layer * 32 + mt0) * 16 + kc) * KS + tap) * 32 + lane) * 16);
            a0.q[0] = p0[0]; a0.q[1] = p0[1];
            const uint4* p1 = (const uint4*)(pw +
                (((((long long)layer * 32 + mt0 + 1) * 16 + kc) * KS + tap) * 32 + lane) * 16);
            a1.q[0] = p1[0]; a1.q[1] = p1[1];
            // B fragments from LDS: lane<16 -> K=0..15, lane>=16 -> K=16..31,
            // row = time index + tap (shifted reuse of staged halo tile).
            const int nn0 = waveN * 32 + llo + tap;
            const uint4* q0 = (const uint4*)(ldsCur + nn0 * LDS_RS + lhi * 16);
            b0.q[0] = q0[0]; b0.q[1] = q0[1];
            const uint4* q1 = (const uint4*)(ldsCur + (nn0 + 16) * LDS_RS + lhi * 16);
            b1.q[0] = q1[0]; b1.q[1] = q1[1];

            acc00 = __builtin_amdgcn_wmma_f32_16x16x32_bf16(false, a0.v, false, b0.v,
                                                            (short)0, acc00, false, false);
            acc01 = __builtin_amdgcn_wmma_f32_16x16x32_bf16(false, a0.v, false, b1.v,
                                                            (short)0, acc01, false, false);
            acc10 = __builtin_amdgcn_wmma_f32_16x16x32_bf16(false, a1.v, false, b0.v,
                                                            (short)0, acc10, false, false);
            acc11 = __builtin_amdgcn_wmma_f32_16x16x32_bf16(false, a1.v, false, b1.v,
                                                            (short)0, acc11, false, false);
        }

        // --- commit next chunk to the opposite LDS buffer ---
        if (kcn < NKC) {
            unsigned short* ldsNxt = lds[kcn & 1];
            #pragma unroll
            for (int s = 0; s < NSTAGE; ++s) {
                int i = tid + s * 256;
                if (i < STAGE_ELEMS) {
                    int k = i / LDS_ROWS;
                    int n = i - k * LDS_ROWS;
                    ldsNxt[n * LDS_RS + k] = reg[s];
                }
            }
        }
        __syncthreads();
    }

    // Epilogue: D layout — element e: M = e + 8*lhi, N = llo. Add bias, store f32.
    const int tBase = t0 + waveN * 32 + llo;
    #pragma unroll
    for (int e = 0; e < 8; ++e) {
        int m0 = mBlk + waveM * 32 + lhi * 8 + e;
        int m1 = m0 + 16;
        float bi0 = bias[layer * FILTER + m0];
        float bi1 = bias[layer * FILTER + m1];
        out[((long long)b * FILTER + m0) * TLEN + tBase]      = acc00[e] + bi0;
        out[((long long)b * FILTER + m0) * TLEN + tBase + 16] = acc01[e] + bi0;
        out[((long long)b * FILTER + m1) * TLEN + tBase]      = acc10[e] + bi1;
        out[((long long)b * FILTER + m1) * TLEN + tBase + 16] = acc11[e] + bi1;
    }
}

// ---------------------------------------------------------------------------
// 4) LayerNorm over channels + exact GELU, fp32 in / bf16 out.
// ---------------------------------------------------------------------------
__global__ void cf_ln_gelu(const float* __restrict__ conv,
                           const float* __restrict__ g,
                           const float* __restrict__ beta,
                           unsigned short* __restrict__ act,
                           int layer) {
    int idx = blockIdx.x * blockDim.x + threadIdx.x;   // b*T + t
    int b = idx >> 12;
    int t = idx & (TLEN - 1);
    const float* src = conv + (long long)b * FILTER * TLEN + t;
    float s = 0.0f, s2 = 0.0f;
    for (int f = 0; f < FILTER; ++f) {
        float v = src[(long long)f * TLEN];
        s += v; s2 += v * v;
    }
    float mu   = s * (1.0f / FILTER);
    float var  = s2 * (1.0f / FILTER) - mu * mu;
    float rstd = rsqrtf(var + 1e-5f);
    unsigned short* dst = act + (long long)b * FILTER * TLEN + t;
    for (int f = 0; f < FILTER; ++f) {
        float v  = (src[(long long)f * TLEN] - mu) * rstd * g[layer * FILTER + f]
                   + beta[layer * FILTER + f];
        float ge = 0.5f * v * (1.0f + erff(v * 0.70710678118654752f));
        dst[(long long)f * TLEN] = f2bf(ge);
    }
}

// ---------------------------------------------------------------------------
// 5) Projection conv (512 -> 58, pointwise), output * mask, fp32.
// ---------------------------------------------------------------------------
__global__ void cf_proj(const unsigned short* __restrict__ act,
                        const float* __restrict__ pwv,
                        const float* __restrict__ pb,
                        const float* __restrict__ mask,
                        float* __restrict__ out) {
    int t = blockIdx.x * blockDim.x + threadIdx.x;
    int f = blockIdx.y;
    int b = blockIdx.z;
    const unsigned short* a = act + (long long)b * FILTER * TLEN + t;
    float acc = pb[f];
    for (int c = 0; c < FILTER; ++c)
        acc += pwv[f * FILTER + c] * bf2f(a[(long long)c * TLEN]);
    out[((long long)b * PROJ_CH + f) * TLEN + t] = acc * mask[b * TLEN + t];
}

// ---------------------------------------------------------------------------
// 6) Rational-quadratic spline + x0 passthrough + per-element logdet terms.
// ---------------------------------------------------------------------------
__global__ void cf_spline(const float* __restrict__ x,
                          const float* __restrict__ mask,
                          const float* __restrict__ hproj,
                          float* __restrict__ out,
                          float* __restrict__ ladBuf) {
    int idx = blockIdx.x * blockDim.x + threadIdx.x;     // b*2*T + c*T + t
    if (idx >= BATCH * HALF * TLEN) return;
    int t = idx & (TLEN - 1);
    int c = (idx >> 12) & 1;
    int b = idx >> 13;
    float m = mask[b * TLEN + t];

    // x0 passthrough channel (masked)
    float x0v = x[((long long)b * C_IN + c) * TLEN + t];
    out[((long long)b * C_IN + c) * TLEN + t] = x0v * m;

    float x1 = x[((long long)b * C_IN + HALF + c) * TLEN + t];
    const float* hp = hproj + ((long long)b * PROJ_CH + c * 29) * TLEN + t;
    const float s = 0.04419417382415922f;                 // 1/sqrt(512)

    float uw[NBINS], uh[NBINS], ud[NBINS - 1];
    #pragma unroll
    for (int j = 0; j < NBINS; ++j) uw[j] = hp[(long long)j * TLEN] * s;
    #pragma unroll
    for (int j = 0; j < NBINS; ++j) uh[j] = hp[(long long)(NBINS + j) * TLEN] * s;
    #pragma unroll
    for (int j = 0; j < NBINS - 1; ++j) ud[j] = hp[(long long)(2 * NBINS + j) * TLEN];

    // widths: softmax -> affine -> cumsum, endpoints pinned to [-5,5]
    float mx = uw[0];
    #pragma unroll
    for (int j = 1; j < NBINS; ++j) mx = fmaxf(mx, uw[j]);
    float ssum = 0.0f, ew[NBINS];
    #pragma unroll
    for (int j = 0; j < NBINS; ++j) { ew[j] = expf(uw[j] - mx); ssum += ew[j]; }
    float inv = 1.0f / ssum;
    float cw[NBINS + 1]; cw[0] = -5.0f;
    float run = 0.0f;
    #pragma unroll
    for (int j = 0; j < NBINS; ++j) {
        float wj = 0.001f + 0.99f * (ew[j] * inv);
        run += wj;
        cw[j + 1] = 10.0f * run - 5.0f;
    }
    cw[NBINS] = 5.0f;

    // heights: same recipe
    mx = uh[0];
    #pragma unroll
    for (int j = 1; j < NBINS; ++j) mx = fmaxf(mx, uh[j]);
    ssum = 0.0f;
    float eh[NBINS];
    #pragma unroll
    for (int j = 0; j < NBINS; ++j) { eh[j] = expf(uh[j] - mx); ssum += eh[j]; }
    inv = 1.0f / ssum;
    float ch[NBINS + 1]; ch[0] = -5.0f;
    run = 0.0f;
    #pragma unroll
    for (int j = 0; j < NBINS; ++j) {
        float hj = 0.001f + 0.99f * (eh[j] * inv);
        run += hj;
        ch[j + 1] = 10.0f * run - 5.0f;
    }
    ch[NBINS] = 5.0f;

    // derivatives: boundary pads give softplus(const)+MIN_D == 1.0 exactly
    float dd[NBINS + 1];
    dd[0] = 1.0f; dd[NBINS] = 1.0f;
    #pragma unroll
    for (int j = 0; j < NBINS - 1; ++j) {
        float u  = ud[j];
        float sp = (u > 20.0f) ? u : log1pf(expf(u));
        dd[j + 1] = 0.001f + sp;
    }

    bool inside = (x1 >= -5.0f) && (x1 <= 5.0f);
    float xin = fminf(fmaxf(x1, -5.0f), 5.0f);
    int bin = -1;
    #pragma unroll
    for (int j = 0; j <= NBINS; ++j) {
        float loc = cw[j] + ((j == NBINS) ? 1e-6f : 0.0f);
        bin += (xin >= loc) ? 1 : 0;
    }
    bin = min(max(bin, 0), NBINS - 1);

    float in_cw = 0.f, in_w = 1.f, in_ch = 0.f, in_h = 1.f, in_d = 1.f, in_d1 = 1.f;
    #pragma unroll
    for (int j = 0; j < NBINS; ++j) {
        if (j == bin) {
            in_cw = cw[j];       in_w  = cw[j + 1] - cw[j];
            in_ch = ch[j];       in_h  = ch[j + 1] - ch[j];
            in_d  = dd[j];       in_d1 = dd[j + 1];
        }
    }
    float in_delta = in_h / in_w;
    float theta = (xin - in_cw) / in_w;
    float tomt  = theta * (1.0f - theta);
    float th2   = theta * theta;
    float num   = in_h * (in_delta * th2 + in_d * tomt);
    float den   = in_delta + (in_d + in_d1 - 2.0f * in_delta) * tomt;
    float o     = in_ch + num / den;
    float omt   = 1.0f - theta;
    float dnum  = in_delta * in_delta * (in_d1 * th2 + 2.0f * in_delta * tomt + in_d * omt * omt);
    float lad   = logf(dnum) - 2.0f * logf(den);

    o   = inside ? o : x1;
    lad = inside ? lad : 0.0f;

    out[((long long)b * C_IN + HALF + c) * TLEN + t] = o * m;
    ladBuf[idx] = lad * m;
}

// ---------------------------------------------------------------------------
// 7) Deterministic per-batch logdet reduction (no atomics).
// ---------------------------------------------------------------------------
__global__ void cf_logdet(const float* __restrict__ lad, float* __restrict__ out) {
    __shared__ float red[256];
    int b = blockIdx.x;
    float s = 0.0f;
    for (int i = threadIdx.x; i < HALF * TLEN; i += 256)
        s += lad[(long long)b * HALF * TLEN + i];
    red[threadIdx.x] = s;
    __syncthreads();
    for (int st = 128; st > 0; st >>= 1) {
        if (threadIdx.x < st) red[threadIdx.x] += red[threadIdx.x + st];
        __syncthreads();
    }
    if (threadIdx.x == 0) out[(long long)BATCH * C_IN * TLEN + b] = red[0];
}

// ---------------------------------------------------------------------------
// Host launch
// ---------------------------------------------------------------------------
extern "C" void kernel_launch(void* const* d_in, const int* in_sizes, int n_in,
                              void* d_out, int out_size, void* d_ws, size_t ws_size,
                              hipStream_t stream) {
    const float* x      = (const float*)d_in[0];
    const float* x_mask = (const float*)d_in[1];
    const float* pre_w  = (const float*)d_in[2];
    const float* pre_b  = (const float*)d_in[3];
    const float* conv_w = (const float*)d_in[4];
    const float* conv_b = (const float*)d_in[5];
    const float* ln_g   = (const float*)d_in[6];
    const float* ln_b   = (const float*)d_in[7];
    const float* proj_w = (const float*)d_in[8];
    const float* proj_b = (const float*)d_in[9];
    float* out = (float*)d_out;

    char* ws = (char*)d_ws;
    size_t off = 0;
    auto take = [&](size_t bytes) -> char* {
        char* p = ws + off;
        off = (off + bytes + 255) & ~(size_t)255;
        return p;
    };
    const size_t nPacked = (size_t)NLAYERS * FILTER * FILTER * KS;        // 5,242,880
    unsigned short* packedW = (unsigned short*)take(nPacked * 2);
    unsigned short* actA    = (unsigned short*)take((size_t)BATCH * FILTER * TLEN * 2);
    float* convOut = (float*)take((size_t)BATCH * FILTER * TLEN * 4);
    float* projOut = (float*)take((size_t)BATCH * PROJ_CH * TLEN * 4);
    float* ladBuf  = (float*)take((size_t)BATCH * HALF * TLEN * 4);

    cf_pack_w<<<(unsigned)(nPacked / 256), 256, 0, stream>>>(conv_w, packedW);
    cf_pre<<<(unsigned)((size_t)BATCH * FILTER * TLEN / 256), 256, 0, stream>>>(
        x, pre_w, pre_b, actA);

    dim3 gConv(TLEN / N_BLK, FILTER / M_BLK, BATCH);     // 64 x 4 x 16
    for (int l = 0; l < NLAYERS; ++l) {
        cf_conv_wmma<<<gConv, 256, 0, stream>>>(actA, packedW, conv_b, x_mask, convOut, l);
        cf_ln_gelu<<<(BATCH * TLEN) / 256, 256, 0, stream>>>(convOut, ln_g, ln_b, actA, l);
    }

    dim3 gProj(TLEN / 256, PROJ_CH, BATCH);
    cf_proj<<<gProj, 256, 0, stream>>>(actA, proj_w, proj_b, x_mask, projOut);

    cf_spline<<<(BATCH * HALF * TLEN) / 256, 256, 0, stream>>>(
        x, x_mask, projOut, out, ladBuf);

    cf_logdet<<<BATCH, 256, 0, stream>>>(ladBuf, out);
}
